// TopkMoE_76845554860267
// MI455X (gfx1250) — matrix operand order repaired and checked by reading, not compiled, not run on hardware
//
#include <hip/hip_runtime.h>
#include <hip/hip_bf16.h>

#define N_TOK   8192
#define DIM     4096
#define NEXP    8
#define RANK    128
#define SCALING 0.25f      // LORA_ALPHA / R = 32 / 128
#define MTILE   32
#define TILES_PER_E (N_TOK / MTILE)   // 256

typedef __attribute__((ext_vector_type(16))) __bf16 v16bf;
typedef __attribute__((ext_vector_type(8)))  float  v8f;

// 16 contiguous floats -> bf16 fragment (one contiguous run; unused)
__device__ __forceinline__ v16bf ld16_cvt(const float* __restrict__ p) {
    const float4* q = (const float4*)p;
    float4 f0 = q[0], f1 = q[1], f2 = q[2], f3 = q[3];
    v16bf r;
    r[0]  = (__bf16)f0.x; r[1]  = (__bf16)f0.y; r[2]  = (__bf16)f0.z; r[3]  = (__bf16)f0.w;
    r[4]  = (__bf16)f1.x; r[5]  = (__bf16)f1.y; r[6]  = (__bf16)f1.z; r[7]  = (__bf16)f1.w;
    r[8]  = (__bf16)f2.x; r[9]  = (__bf16)f2.y; r[10] = (__bf16)f2.z; r[11] = (__bf16)f2.w;
    r[12] = (__bf16)f3.x; r[13] = (__bf16)f3.y; r[14] = (__bf16)f3.z; r[15] = (__bf16)f3.w;
    return r;
}

// two contiguous 8-float runs -> bf16 fragment (A-fragment interleave)
__device__ __forceinline__ v16bf ld8x2_cvt(const float* __restrict__ p0,
                                           const float* __restrict__ p1) {
    const float4* q0 = (const float4*)p0;
    const float4* q1 = (const float4*)p1;
    float4 f0 = q0[0], f1 = q0[1], f2 = q1[0], f3 = q1[1];
    v16bf r;
    r[0]  = (__bf16)f0.x; r[1]  = (__bf16)f0.y; r[2]  = (__bf16)f0.z; r[3]  = (__bf16)f0.w;
    r[4]  = (__bf16)f1.x; r[5]  = (__bf16)f1.y; r[6]  = (__bf16)f1.z; r[7]  = (__bf16)f1.w;
    r[8]  = (__bf16)f2.x; r[9]  = (__bf16)f2.y; r[10] = (__bf16)f2.z; r[11] = (__bf16)f2.w;
    r[12] = (__bf16)f3.x; r[13] = (__bf16)f3.y; r[14] = (__bf16)f3.z; r[15] = (__bf16)f3.w;
    return r;
}

// ---------------------------------------------------------------------------
// Kernel 0: zero the output and the per-expert counters (graph-replay safe).
// ---------------------------------------------------------------------------
__global__ void zero_init_kernel(float* __restrict__ out, int* __restrict__ counts,
                                 long total) {
    long idx    = (long)blockIdx.x * blockDim.x + threadIdx.x;
    long stride = (long)gridDim.x * blockDim.x;
    for (long i = idx; i < total; i += stride) out[i] = 0.0f;
    if (idx < NEXP) counts[idx] = 0;
}

// ---------------------------------------------------------------------------
// Kernel 1: router. One block (256 threads) per token.
// ---------------------------------------------------------------------------
__global__ __launch_bounds__(256)
void router_kernel(const float* __restrict__ x,
                   const float* __restrict__ Wr,
                   const float* __restrict__ br,
                   int*   __restrict__ counts,
                   int*   __restrict__ tokList,
                   float* __restrict__ wgtList) {
    __shared__ float red[NEXP][256];
    const int n   = blockIdx.x;
    const int tid = threadIdx.x;

    float acc[NEXP];
#pragma unroll
    for (int e = 0; e < NEXP; ++e) acc[e] = 0.0f;

    const float* xr = x + (long)n * DIM;
    for (int i = tid; i < DIM; i += 256) {
        float xv = xr[i];
#pragma unroll
        for (int e = 0; e < NEXP; ++e) acc[e] += xv * Wr[e * DIM + i];
    }
#pragma unroll
    for (int e = 0; e < NEXP; ++e) red[e][tid] = acc[e];
    __syncthreads();
    for (int off = 128; off > 0; off >>= 1) {
        if (tid < off) {
#pragma unroll
            for (int e = 0; e < NEXP; ++e) red[e][tid] += red[e][tid + off];
        }
        __syncthreads();
    }

    if (tid == 0) {
        float lg[NEXP];
        float mx = -3.0e38f;
#pragma unroll
        for (int e = 0; e < NEXP; ++e) { lg[e] = red[e][0] + br[e]; mx = fmaxf(mx, lg[e]); }
        float p[NEXP]; float s = 0.0f;
#pragma unroll
        for (int e = 0; e < NEXP; ++e) { p[e] = __expf(lg[e] - mx); s += p[e]; }
        float inv = 1.0f / s;
#pragma unroll
        for (int e = 0; e < NEXP; ++e) p[e] *= inv;
        int i0 = 0;
#pragma unroll
        for (int e = 1; e < NEXP; ++e) if (p[e] > p[i0]) i0 = e;
        int i1 = (i0 == 0) ? 1 : 0;
#pragma unroll
        for (int e = 0; e < NEXP; ++e) if (e != i0 && p[e] > p[i1]) i1 = e;
        float rsum = 1.0f / (p[i0] + p[i1]);

        int pos0 = atomicAdd(&counts[i0], 1);
        tokList[i0 * N_TOK + pos0] = n;
        wgtList[i0 * N_TOK + pos0] = p[i0] * rsum * SCALING;
        int pos1 = atomicAdd(&counts[i1], 1);
        tokList[i1 * N_TOK + pos1] = n;
        wgtList[i1 * N_TOK + pos1] = p[i1] * rsum * SCALING;
    }
}

// ---------------------------------------------------------------------------
// Kernel 2: per-expert fused LoRA GEMM, bf16 WMMA, fragments built directly
// from global memory with 128-bit loads (no LDS staging in the K loop).
// Block = 32 gathered tokens of one expert, 256 threads (8 wave32).
// Stage 1: H[32,128] = Xg @ A_e^T      (wave w owns R columns [16w,16w+16))
// Stage 2: out += w_tok * (H @ B_e^T)  (wave w owns 32 of 256 D-tiles;
//          all stage-2 A-fragments hoisted into VGPRs before the loop)
// ---------------------------------------------------------------------------
__global__ __launch_bounds__(256)
void moe_wmma_kernel(const float* __restrict__ x,
                     const float* __restrict__ A,
                     const float* __restrict__ B,
                     const int*   __restrict__ counts,
                     const int*   __restrict__ tokList,
                     const float* __restrict__ wgtList,
                     float* __restrict__ out) {
    __shared__ __bf16 sH[MTILE * RANK];  // H intermediate [m][r]
    __shared__ float  sW[MTILE];
    __shared__ int    sTok[MTILE];

    const int e    = blockIdx.x / TILES_PER_E;
    const int tile = blockIdx.x % TILES_PER_E;
    const int cnt  = counts[e];
    if (tile * MTILE >= cnt) return;     // uniform early-exit (before barriers)

    const int tid  = threadIdx.x;
    const int lane = tid & 31;
    const int w    = tid >> 5;

    if (tid < MTILE) {
        int row = tile * MTILE + tid;
        if (row < cnt) {
            sTok[tid] = tokList[e * N_TOK + row];
            sW[tid]   = wgtList[e * N_TOK + row];
        } else {
            sTok[tid] = 0;       // padded row: weight 0 -> contributes nothing
            sW[tid]   = 0.0f;
        }
    }
    __syncthreads();

    const int ln  = lane & 15;           // frag row (A) / column (B,D)
    const int hi  = (lane >> 4) & 1;     // lane-half selector
    const int kbA = hi * 8;              // A-frag K base within 16-run pattern
    const int kbB = hi * 16;             // unused: B-frags use kbA, same K order as A
    const int mD  = hi * 8;              // D-frag M base

    const float* Abase = A + (long)e * RANK * DIM;
    const float* Bbase = B + (long)e * DIM * RANK;

    // per-lane row pointers (constant across the K loop)
    const float* xrow0 = x + (long)sTok[ln]      * DIM;   // M-subtile 0
    const float* xrow1 = x + (long)sTok[16 + ln] * DIM;   // M-subtile 1
    const float* arow  = Abase + (long)(w * 16 + ln) * DIM;

    v8f accH0 = {}, accH1 = {};

    // ---------------- Stage 1: K loop over D (no barriers) ----------------
    for (int k0 = 0; k0 < DIM; k0 += 32) {
        v16bf bfrag = ld8x2_cvt(arow + k0 + kbA, arow + k0 + 16 + kbA);
        v16bf a0 = ld8x2_cvt(xrow0 + k0 + kbA, xrow0 + k0 + 16 + kbA);
        v16bf a1 = ld8x2_cvt(xrow1 + k0 + kbA, xrow1 + k0 + 16 + kbA);
        accH0 = __builtin_amdgcn_wmma_f32_16x16x32_bf16(
            false, a0, false, bfrag, (short)0, accH0, false, false);
        accH1 = __builtin_amdgcn_wmma_f32_16x16x32_bf16(
            false, a1, false, bfrag, (short)0, accH1, false, false);
        if (k0 + 32 < DIM)
            __builtin_prefetch(arow + k0 + 32, 0, 1);
    }

    // spill H to LDS as bf16 (single handoff barrier)
#pragma unroll
    for (int v = 0; v < 8; ++v) {
        sH[(v + mD) * RANK      + w * 16 + ln] = (__bf16)accH0[v];
        sH[(16 + v + mD) * RANK + w * 16 + ln] = (__bf16)accH1[v];
    }
    __syncthreads();

    // ---------------- Stage 2 prologue: hoist loop-invariants ----------------
    const __bf16* h0 = &sH[ln * RANK];
    const __bf16* h1 = &sH[(16 + ln) * RANK];

    v16bf af0[4], af1[4];                // stage-2 A-fragments (kk = q*32)
#pragma unroll
    for (int q = 0; q < 4; ++q) {
        const int kk = q * 32;
#pragma unroll
        for (int j = 0; j < 8; ++j) {
            af0[q][j]     = h0[kk + kbA + j];
            af0[q][j + 8] = h0[kk + 16 + kbA + j];
            af1[q][j]     = h1[kk + kbA + j];
            af1[q][j + 8] = h1[kk + 16 + kbA + j];
        }
    }

    float  wv0[8], wv1[8];
    float* op0[8]; float* op1[8];
#pragma unroll
    for (int v = 0; v < 8; ++v) {
        const int m0 = v + mD, m1 = 16 + v + mD;
        wv0[v] = sW[m0];
        wv1[v] = sW[m1];
        op0[v] = out + (long)sTok[m0] * DIM + ln;
        op1[v] = out + (long)sTok[m1] * DIM + ln;
    }

    // ---------------- Stage 2: N loop over D, K = RANK ----------------
    for (int t = 0; t < 32; ++t) {
        const int dcol0 = (w * 32 + t) * 16;
        const float* brow = Bbase + (long)(dcol0 + ln) * RANK;
        if (t + 1 < 32)
            __builtin_prefetch(brow + 16 * RANK, 0, 1);   // next D-tile's B row

        v8f y0 = {}, y1 = {};
#pragma unroll
        for (int q = 0; q < 4; ++q) {
            v16bf b2 = ld8x2_cvt(brow + q * 32 + kbA, brow + q * 32 + 16 + kbA);
            y0 = __builtin_amdgcn_wmma_f32_16x16x32_bf16(
                false, af0[q], false, b2, (short)0, y0, false, false);
            y1 = __builtin_amdgcn_wmma_f32_16x16x32_bf16(
                false, af1[q], false, b2, (short)0, y1, false, false);
        }
#pragma unroll
        for (int v = 0; v < 8; ++v) {
            atomicAdd(op0[v] + dcol0, y0[v] * wv0[v]);
            atomicAdd(op1[v] + dcol0, y1[v] * wv1[v]);
        }
    }
}

// ---------------------------------------------------------------------------
// Launcher
// ---------------------------------------------------------------------------
extern "C" void kernel_launch(void* const* d_in, const int* in_sizes, int n_in,
                              void* d_out, int out_size, void* d_ws, size_t ws_size,
                              hipStream_t stream) {
    const float* x  = (const float*)d_in[0];
    const float* Wr = (const float*)d_in[1];
    const float* br = (const float*)d_in[2];
    const float* A  = (const float*)d_in[3];
    const float* B  = (const float*)d_in[4];
    float* out = (float*)d_out;

    char* ws = (char*)d_ws;
    int*   counts  = (int*)ws;                                   // 8 ints (pad 256B)
    int*   tokList = (int*)(ws + 256);                           // E*N ints
    float* wgtList = (float*)(ws + 256 + (size_t)NEXP * N_TOK * sizeof(int));

    zero_init_kernel<<<2048, 256, 0, stream>>>(out, counts, (long)N_TOK * DIM);
    router_kernel<<<N_TOK, 256, 0, stream>>>(x, Wr, br, counts, tokList, wgtList);
    moe_wmma_kernel<<<NEXP * TILES_PER_E, 256, 0, stream>>>(
        x, A, B, counts, tokList, wgtList, out);
}
